// ColumnParallelLinearWithDelta_28973849379099
// MI455X (gfx1250) — matrix operand order repaired and checked by reading, not compiled, not run on hardware
//
#include <hip/hip_runtime.h>
#include <stdint.h>

#define T_TOK 8192
#define K_DIM 4096
#define N_DIM 4096
#define D_NUM 4
#define KWORDS (K_DIM / 8)
#define NW8    (N_DIM / 8)

#define BM 128
#define BN 128
#define BK 64
#define LDSS 72              // ushorts per LDS row: 64 data + 8 pad (144B, bank-rotating)

// ---- CDNA5 feature detection (device pass only; host pass falls back) ----
#if defined(__has_builtin)
#  if __has_builtin(__builtin_amdgcn_global_load_async_to_lds_b128) && \
      __has_builtin(__builtin_amdgcn_s_wait_asynccnt)
#    define USE_ASYNC 1
#  endif
#  if __has_builtin(__builtin_amdgcn_tensor_load_to_lds) && \
      __has_builtin(__builtin_amdgcn_s_wait_tensorcnt)
#    define USE_TDM 1
#  endif
#endif
#ifndef USE_ASYNC
#define USE_ASYNC 0
#endif
#ifndef USE_TDM
#define USE_TDM 0
#endif

typedef __attribute__((ext_vector_type(16))) __bf16 v16bf;
typedef __attribute__((ext_vector_type(8)))  float  v8f;

union Frag { uint4 u[2]; v16bf v; };
union AccU { v8f v; float f[8]; };

__device__ __forceinline__ unsigned int f2bf(float f) {
    unsigned int u = __float_as_uint(f);
    u += 0x7FFFu + ((u >> 16) & 1u);      // round-to-nearest-even
    return u >> 16;
}

__device__ __forceinline__ unsigned lds_off32(const void* p) {
    return (unsigned)(size_t)p;           // LDS aperture: low 32 bits = LDS offset
}

#if USE_ASYNC
typedef __attribute__((ext_vector_type(4))) int v4i_t;
typedef __attribute__((address_space(1))) v4i_t* g_v4i_p;   // global src (AS1)
typedef __attribute__((address_space(3))) v4i_t* l_v4i_p;   // LDS dst (AS3)
#endif

// 16B global -> LDS copy; async engine when available
__device__ __forceinline__ void cp16(const void* g, void* l) {
#if USE_ASYNC
    __builtin_amdgcn_global_load_async_to_lds_b128(
        (g_v4i_p)(size_t)g, (l_v4i_p)(size_t)lds_off32(l), 0, 0);
#else
    *(uint4*)l = *(const uint4*)g;
#endif
}

__device__ __forceinline__ void wait_async_done() {
#if USE_ASYNC
    __builtin_amdgcn_s_wait_asynccnt(0);
#endif
}

#if USE_TDM
typedef unsigned int tdm_v4u __attribute__((ext_vector_type(4)));
typedef int          tdm_v8i __attribute__((ext_vector_type(8)));
typedef int          tdm_v4i __attribute__((ext_vector_type(4)));

// TDM: load a 128-row x 64-elem bf16 tile (row stride K) into LDS with
// 16B padding every 128B row (pad_interval=32 dwords, pad_amount=4 dwords)
__device__ __forceinline__ void tdm_load_b_tile(const void* gsrc, unsigned ldsOff) {
    unsigned long long ga = (unsigned long long)(size_t)gsrc;
    tdm_v4u g0 = { 1u,                                     // count=1, user descriptor
                   ldsOff,                                 // lds_addr
                   (unsigned)ga,                           // global_addr[31:0]
                   ((unsigned)(ga >> 32) & 0x01FFFFFFu) | 0x80000000u }; // addr[56:32] | type=2
    tdm_v8i g1 = { 0x07110000,                             // pad_amt=3(16B), pad_int=4(128B), pad_en, data_size=2B
                   (int)(K_DIM << 16),                     // tensor_dim0[15:0]<<16 (4096)
                   (int)(N_DIM << 16),                     // tdim0 hi | tensor_dim1[15:0]<<16
                   (int)(BK << 16),                        // tdim1 hi | tile_dim0=64
                   BM,                                     // tile_dim1=128 | tile_dim2=0
                   K_DIM,                                  // tensor_dim0_stride lo (4096)
                   0, 0 };
    tdm_v4i g2 = { 0, 0, 0, 0 };                           // 2-D tile: groups 2/3 unused
    tdm_v4i g3 = { 0, 0, 0, 0 };
    tdm_v8i g4 = { 0, 0, 0, 0, 0, 0, 0, 0 };               // clang-23 6-arg form
    __builtin_amdgcn_tensor_load_to_lds(g0, g1, g2, g3, g4, 0);
}
#endif

// ---------------- routing: histogram + scatter (token sort by delta id) ---------
__global__ void route_zero_kernel(int* hdr) {
    if (threadIdx.x < 8) hdr[threadIdx.x] = 0;
}
__global__ void route_hist_kernel(const int* __restrict__ indices, int* hdr) {
    int t = blockIdx.x * 256 + threadIdx.x;
    if (t < T_TOK) atomicAdd(&hdr[indices[t]], 1);
}
__global__ void route_scatter_kernel(const int* __restrict__ indices, int* hdr,
                                     int* __restrict__ perm) {
    int t = blockIdx.x * 256 + threadIdx.x;
    if (t >= T_TOK) return;
    int d = indices[t];
    int start = 0;
    #pragma unroll
    for (int e = 0; e < D_NUM; ++e) if (e < d) start += hdr[e];
    int pos = start + atomicAdd(&hdr[4 + d], 1);
    perm[pos] = t;
}

// ---------------- fp32 -> bf16 conversion of activations --------------------
__global__ void cvt_bf16_kernel(const float* __restrict__ src,
                                unsigned short* __restrict__ dst, int n) {
    int i = (blockIdx.x * 256 + threadIdx.x) * 8;
    if (i >= n) return;
    float4 a = *(const float4*)(src + i);
    float4 b = *(const float4*)(src + i + 4);
    uint4 o;
    o.x = f2bf(a.x) | (f2bf(a.y) << 16);
    o.y = f2bf(a.z) | (f2bf(a.w) << 16);
    o.z = f2bf(b.x) | (f2bf(b.y) << 16);
    o.w = f2bf(b.z) | (f2bf(b.w) << 16);
    *(uint4*)(dst + i) = o;
}

// -------- wbd[d][n][k] = bf16( W[n][k] + scale[d][n]*(q4[d][k][n] - z[d][n]) ) --
// Dequant happens exactly once per element here instead of per GEMM block.
#define WBD_NT 64
#define WBD_KW 8
__global__ __launch_bounds__(256)
void build_wbd_kernel(const float* __restrict__ weight,
                      const int* __restrict__ qweight,
                      const int* __restrict__ qzeros,
                      const float* __restrict__ scales,
                      unsigned short* __restrict__ wbd) {
    __shared__ int   lsQ[WBD_KW * WBD_NT];
    __shared__ float lsS[WBD_NT];
    __shared__ float lsZ[WBD_NT];
    const int d = blockIdx.z;
    const int nBase = blockIdx.y * WBD_NT;
    const int kwBase = blockIdx.x * WBD_KW;
    const int tid = threadIdx.x;
    // phase 1: coalesced qweight fetch (n fastest across lanes)
    #pragma unroll
    for (int it = 0; it < 2; ++it) {
        int idx = tid + it * 256;            // 0..511
        int n = idx & 63, kw = idx >> 6;
        lsQ[kw * WBD_NT + n] =
            qweight[((size_t)d * KWORDS + kwBase + kw) * N_DIM + nBase + n];
    }
    if (tid < WBD_NT) {
        int n = nBase + tid;
        lsS[tid] = scales[d * N_DIM + n];
        int zw = qzeros[d * NW8 + (n >> 3)];
        lsZ[tid] = (float)((zw >> ((n & 7) * 4)) & 0xF);
    }
    __syncthreads();
    // phase 2: k-fastest combine, coalesced W read + wbd write (LDS transpose)
    #pragma unroll
    for (int it = 0; it < 2; ++it) {
        int idx = tid + it * 256;
        int kw = idx & 7, n = idx >> 3;      // n 0..63
        int gn = nBase + n;
        int gk = (kwBase + kw) * 8;
        const float* wsrc = weight + (size_t)gn * K_DIM + gk;
        float4 a = *(const float4*)wsrc;
        float4 b = *(const float4*)(wsrc + 4);
        int qw = lsQ[kw * WBD_NT + n];       // one i32 = these exact 8 k nibbles
        float s = lsS[n], z = lsZ[n];
        float f[8] = {a.x, a.y, a.z, a.w, b.x, b.y, b.z, b.w};
        uint4 o; unsigned* po = &o.x;
        #pragma unroll
        for (int p = 0; p < 4; ++p) {
            float v0 = f[2*p]   + s * ((float)((qw >> (8*p))     & 0xF) - z);
            float v1 = f[2*p+1] + s * ((float)((qw >> (8*p + 4)) & 0xF) - z);
            po[p] = f2bf(v0) | (f2bf(v1) << 16);
        }
        *(uint4*)(wbd + ((size_t)d * N_DIM + gn) * K_DIM + gk) = o;
    }
}

// ---------------- pure-bf16 fused GEMM: out = gather(x) @ wbd[d]^T + bias ------
__global__ __launch_bounds__(256, 2)
void fused_delta_gemm_kernel(const unsigned short* __restrict__ xb,
                             const unsigned short* __restrict__ wbd,
                             const float* __restrict__ bias,
                             const int* __restrict__ hdr,
                             const int* __restrict__ perm,
                             float* __restrict__ out) {
    __shared__ __align__(16) unsigned short lsA[2][BM * LDSS];
    __shared__ __align__(16) unsigned short lsB[2][BN * LDSS];
    __shared__ int lsPerm[BM];

    const int tid = threadIdx.x;

    // ---- decode M tile -> (delta id, row range in sorted perm) ----
    int cnt[D_NUM];
    #pragma unroll
    for (int g = 0; g < D_NUM; ++g) cnt[g] = hdr[g];
    int m = blockIdx.y;
    int d = -1, rowStart = 0, validRows = 0;
    {
        int tileAcc = 0, gStart = 0;
        #pragma unroll
        for (int g = 0; g < D_NUM; ++g) {
            int tiles = (cnt[g] + BM - 1) / BM;
            if (d < 0 && m < tileAcc + tiles) {
                d = g;
                int tIn = m - tileAcc;
                rowStart  = gStart + tIn * BM;
                validRows = min(BM, cnt[g] - tIn * BM);
            }
            tileAcc += tiles;
            gStart  += cnt[g];
        }
    }
    if (d < 0) return;
    const int colBase = blockIdx.x * BN;

    if (tid < BM)
        lsPerm[tid] = (tid < validRows) ? perm[rowStart + tid] : perm[rowStart];
    __syncthreads();

    const int lane  = tid & 31;
    const int wave  = tid >> 5;
    const int waveM = wave >> 1;
    const int waveN = wave & 1;
    const int r16   = lane & 15;
    const int hi    = lane >> 4;

    // per-thread fixed staging slots (row gather pointers hoisted out of k-loop)
    const unsigned short* aSrc[4];
    int aOff[4];
    #pragma unroll
    for (int it = 0; it < 4; ++it) {
        int chunk = tid + it * 256;          // 1024 x 16B chunks
        int row = chunk >> 3, c8 = chunk & 7;
        aSrc[it] = xb + (size_t)lsPerm[row] * K_DIM + c8 * 8;
        aOff[it] = row * LDSS + c8 * 8;
    }
    const unsigned short* bBase = wbd + ((size_t)d * N_DIM + colBase) * K_DIM;
#if !USE_TDM
    const unsigned short* bSrc[4];
    int bOff[4];
    #pragma unroll
    for (int it = 0; it < 4; ++it) {
        int chunk = tid + it * 256;
        int nloc = chunk >> 3, c8 = chunk & 7;
        bSrc[it] = bBase + (size_t)nloc * K_DIM + c8 * 8;
        bOff[it] = nloc * LDSS + c8 * 8;
    }
#endif

    v8f acc[2][4] = {};

    // ---- prologue: fill buffer 0 ----
    #pragma unroll
    for (int it = 0; it < 4; ++it) cp16(aSrc[it], &lsA[0][aOff[it]]);
#if USE_TDM
    if (wave == 0) tdm_load_b_tile(bBase, lds_off32(&lsB[0][0]));
#else
    #pragma unroll
    for (int it = 0; it < 4; ++it) cp16(bSrc[it], &lsB[0][bOff[it]]);
#endif

    for (int kc = 0; kc < K_DIM; kc += BK) {
        const int buf = (kc / BK) & 1;

        // wait for buffer `buf` (own async loads; wave0 owns the TDM op), then rendezvous
        wait_async_done();
#if USE_TDM
        if (wave == 0) __builtin_amdgcn_s_wait_tensorcnt(0);
#endif
        __syncthreads();

        // kick off next buffer while we compute this one
        if (kc + BK < K_DIM) {
            const int nb = buf ^ 1;
            const int kn = kc + BK;
            #pragma unroll
            for (int it = 0; it < 4; ++it) cp16(aSrc[it] + kn, &lsA[nb][aOff[it]]);
#if USE_TDM
            if (wave == 0) tdm_load_b_tile(bBase + kn, lds_off32(&lsB[nb][0]));
#else
            #pragma unroll
            for (int it = 0; it < 4; ++it) cp16(bSrc[it] + kn, &lsB[nb][bOff[it]]);
#endif
        }

        // ---- WMMA over the staged 128x64 x 64x128 tile ----
        #pragma unroll
        for (int ks = 0; ks < BK; ks += 32) {
            Frag a[2], b[4];
            #pragma unroll
            for (int mi = 0; mi < 2; ++mi) {
                // A 16x32 bf16: lane<16 -> row r16, K {0..7,16..23}; lane>=16 -> K {8..15,24..31}
                const unsigned short* p =
                    &lsA[buf][(waveM * 32 + mi * 16 + r16) * LDSS + ks + hi * 8];
                a[mi].u[0] = *(const uint4*)p;
                a[mi].u[1] = *(const uint4*)(p + 16);
            }
            #pragma unroll
            for (int ni = 0; ni < 4; ++ni) {
                // B 32x16 bf16: lane<16 -> col r16, K {0..7,16..23}; lane>=16 -> K {8..15,24..31}
                const unsigned short* p =
                    &lsB[buf][(waveN * 64 + ni * 16 + r16) * LDSS + ks + hi * 8];
                b[ni].u[0] = *(const uint4*)p;
                b[ni].u[1] = *(const uint4*)(p + 16);
            }
            #pragma unroll
            for (int mi = 0; mi < 2; ++mi)
                #pragma unroll
                for (int ni = 0; ni < 4; ++ni)
                    acc[mi][ni] = __builtin_amdgcn_wmma_f32_16x16x32_bf16(
                        false, a[mi].v, false, b[ni].v,
                        (short)0, acc[mi][ni], false, false);
        }
        __syncthreads();
    }

    // ---- epilogue: scatter rows back to token order, add bias ----
    #pragma unroll
    for (int mi = 0; mi < 2; ++mi) {
        #pragma unroll
        for (int ni = 0; ni < 4; ++ni) {
            int col = colBase + waveN * 64 + ni * 16 + r16;
            float bv = bias[col];
            AccU u; u.v = acc[mi][ni];
            #pragma unroll
            for (int e = 0; e < 8; ++e) {
                int rowInTile = waveM * 32 + mi * 16 + hi * 8 + e;
                if (rowInTile < validRows) {
                    int tok = lsPerm[rowInTile];
                    out[(size_t)tok * N_DIM + col] = u.f[e] + bv;
                }
            }
        }
    }
}

extern "C" void kernel_launch(void* const* d_in, const int* in_sizes, int n_in,
                              void* d_out, int out_size, void* d_ws, size_t ws_size,
                              hipStream_t stream) {
    const float* x       = (const float*)d_in[0];
    const float* weight  = (const float*)d_in[1];
    const float* bias    = (const float*)d_in[2];
    const int*   qweight = (const int*)d_in[3];
    const int*   qzeros  = (const int*)d_in[4];
    const float* scales  = (const float*)d_in[5];
    const int*   indices = (const int*)d_in[6];
    float*       out     = (float*)d_out;

    uint8_t* ws = (uint8_t*)d_ws;
    int* hdr  = (int*)ws;                                 // counts[4] + cursors[4]
    int* perm = (int*)(ws + 64);                          // 8192 ints
    unsigned short* xb  = (unsigned short*)(ws + 65536);  // x in bf16   (64 MB)
    unsigned short* wbd = xb + (size_t)T_TOK * K_DIM;     // W+delta bf16 (128 MB)

    route_zero_kernel<<<1, 32, 0, stream>>>(hdr);
    route_hist_kernel<<<(T_TOK + 255) / 256, 256, 0, stream>>>(indices, hdr);
    route_scatter_kernel<<<(T_TOK + 255) / 256, 256, 0, stream>>>(indices, hdr, perm);

    cvt_bf16_kernel<<<(T_TOK * K_DIM) / 2048, 256, 0, stream>>>(x, xb, T_TOK * K_DIM);

    dim3 wgrid(K_DIM / (WBD_KW * 8), N_DIM / WBD_NT, D_NUM);
    build_wbd_kernel<<<wgrid, 256, 0, stream>>>(weight, qweight, qzeros, scales, wbd);

    dim3 grid(N_DIM / BN, T_TOK / BM + D_NUM);
    fused_delta_gemm_kernel<<<grid, 256, 0, stream>>>(xb, wbd, bias, hdr, perm, out);
}
